// UniTransformer_29618094473619
// MI455X (gfx1250) — matrix operand order repaired and checked by reading, not compiled, not run on hardware
//
#include <hip/hip_runtime.h>
#include <cstdint>

typedef _Float16 f16;
typedef __attribute__((ext_vector_type(16))) _Float16 v16h;
typedef __attribute__((ext_vector_type(8)))  _Float16 h8;
typedef __attribute__((ext_vector_type(8)))  float    v8f;

#define NN   4000
#define KNN  32
#define NE   (NN*KNN)
#define HID  128
#define KVP  352            // padded KV feature dim (11 * 32)
#define INV_SQRT_DH 0.35355339059327f

struct MlpW { const float *b1,*b2,*bln,*g; const f16 *w1p,*w2p; };

// ---------------- WMMA helpers ----------------

__device__ __forceinline__ v16h load_a(const f16* p16, int lane) {
  // A fragment (16x32 f16): lane<16 holds K{0..15}, lane>=16 holds K{16..31}
  const f16* p = p16 + ((lane & 16) ? 16 : 0);
  union { v16h v; h8 h[2]; } u;
  u.h[0] = *(const h8*)(p);
  u.h[1] = *(const h8*)(p + 8);
  return u.v;
}

// Y[128 x 128] (f32, LDS) = A[128 x KT*32] (f16, LDS, row stride LDA) @ Bp (packed) + b1
template<int KT, int LDA>
__device__ __forceinline__ void gemm_to_y(const f16* A, const f16* Bp, const float* b1,
                                          float* Y, int wave, int lane) {
  v8f acc[8];
#pragma unroll
  for (int nt = 0; nt < 8; ++nt) acc[nt] = (v8f)0.0f;
  const f16* arow = A + (size_t)(wave*16 + (lane & 15)) * LDA;
#pragma unroll
  for (int kt = 0; kt < KT; ++kt) {
    v16h a = load_a(arow + kt*32, lane);
#pragma unroll
    for (int nt = 0; nt < 8; ++nt) {
      v16h b = *(const v16h*)(Bp + (size_t)(kt*8 + nt)*512 + lane*16);
      acc[nt] = __builtin_amdgcn_wmma_f32_16x16x32_f16(false, a, false, b,
                                                       (short)0, acc[nt], false, false);
    }
  }
  const int colb = lane & 15;
  const int rowo = wave*16 + ((lane & 16) ? 8 : 0);
#pragma unroll
  for (int nt = 0; nt < 8; ++nt) {
    int col = nt*16 + colb;
    float bb = b1[col];
#pragma unroll
    for (int i = 0; i < 8; ++i)
      Y[(size_t)(rowo + i)*128 + col] = acc[nt][i] + bb;
  }
}

// acc[8] = Z[128 x 128] (f16 packed over Y rows, stride 256 halves) @ Bp (packed 128x128)
__device__ __forceinline__ void gemm2_128(const f16* Z, const f16* Bp, v8f* acc,
                                          int wave, int lane) {
#pragma unroll
  for (int nt = 0; nt < 8; ++nt) acc[nt] = (v8f)0.0f;
  const f16* zrow = Z + (size_t)(wave*16 + (lane & 15)) * 256;
#pragma unroll
  for (int kt = 0; kt < 4; ++kt) {
    v16h a = load_a(zrow + kt*32, lane);
#pragma unroll
    for (int nt = 0; nt < 8; ++nt) {
      v16h b = *(const v16h*)(Bp + (size_t)(kt*8 + nt)*512 + lane*16);
      acc[nt] = __builtin_amdgcn_wmma_f32_16x16x32_f16(false, a, false, b,
                                                       (short)0, acc[nt], false, false);
    }
  }
}

// LayerNorm(128) + ReLU over Y rows, repacked as f16 in place (row stride 256 halves)
__device__ __forceinline__ void ln_relu_pack(float* Y, const float* g, const float* bln,
                                             int tid) {
  if (tid < 128) {
    float* row = Y + (size_t)tid * 128;
    float mu = 0.f;
    for (int c = 0; c < 128; ++c) mu += row[c];
    mu *= (1.f/128.f);
    float var = 0.f;
    for (int c = 0; c < 128; ++c) { float d = row[c] - mu; var += d*d; }
    var *= (1.f/128.f);
    float rstd = rsqrtf(var + 1e-5f);
    f16* z = (f16*)row;
    for (int c = 0; c < 128; ++c) {
      float v = (row[c] - mu) * rstd * g[c] + bln[c];
      z[c] = (f16)(v > 0.f ? v : 0.f);   // byte 2c < byte 4c: safe in-place repack
    }
  }
}

__device__ __forceinline__ void rbf_consts(float* s_off, float* s_coeff, int tid) {
  if (tid < 20) {
    float step = logf(11.0f) / 19.0f;
    float oi = expf(step * tid) - 1.0f;
    float oo = expf(step * (tid == 0 ? 1 : tid - 1)) - 1.0f;
    float wid = (tid == 0) ? (oo - oi) : (oi - oo);
    s_off[tid]   = oi;
    s_coeff[tid] = -0.5f / (wid * wid);
  }
}

// ---------------- weight packing ----------------
// B fragment layout: tile (kt,nt) -> 512 halves; idx = lane*16+j ;
// n = nt*16 + lane%16 ; k = kt*32 + (lane>=16 ? 16:0) + j

__global__ void pack_plain(const float* __restrict__ W, f16* __restrict__ out,
                           int K, int Kpad, int N) {
  int total = (Kpad/32) * (N/16) * 512;
  for (int t = blockIdx.x*blockDim.x + threadIdx.x; t < total; t += gridDim.x*blockDim.x) {
    int tile = t >> 9, r = t & 511;
    int lane = r >> 4, j = r & 15;
    int ntn = N >> 4;
    int kt = tile / ntn, nt = tile % ntn;
    int n = nt*16 + (lane & 15);
    int k = kt*32 + ((lane & 16) ? 16 : 0) + j;
    out[t] = (f16)((k < K) ? W[(size_t)k*N + n] : 0.f);
  }
}

// KV weights (K=340) remapped to padded 352 rows: [0..83]->[0..83], [84..339]->[88..343]
__global__ void pack_kv(const float* __restrict__ W, f16* __restrict__ out, int N) {
  int total = (KVP/32) * (N/16) * 512;
  for (int t = blockIdx.x*blockDim.x + threadIdx.x; t < total; t += gridDim.x*blockDim.x) {
    int tile = t >> 9, r = t & 511;
    int lane = r >> 4, j = r & 15;
    int ntn = N >> 4;
    int kt = tile / ntn, nt = tile % ntn;
    int n = nt*16 + (lane & 15);
    int k = kt*32 + ((lane & 16) ? 16 : 0) + j;     // padded row index
    int kr; bool ok;
    if (k < 84)                { kr = k;     ok = true; }
    else if (k >= 88 && k < 344) { kr = k - 4; ok = true; }
    else                       { kr = 0;     ok = false; }
    out[t] = (f16)(ok ? W[(size_t)kr*N + n] : 0.f);
  }
}

// ---------------- init kernels ----------------

__global__ void init_state(const float* __restrict__ x, const float* __restrict__ h,
                           float* __restrict__ XA, float* __restrict__ HA) {
  for (int t = blockIdx.x*blockDim.x + threadIdx.x; t < NN*HID; t += gridDim.x*blockDim.x) {
    HA[t] = h[t];
    if (t < NN*4) {
      int n = t >> 2, a = t & 3;
      XA[t] = (a < 3) ? x[n*3 + a] : 0.f;
    }
  }
}

__global__ void init_edges(const float* __restrict__ x, const int* __restrict__ lig,
                           const int* __restrict__ eidx, const float* __restrict__ ew_w,
                           const float* __restrict__ ew_b,
                           float* __restrict__ EW, int* __restrict__ ET) {
  for (int t = blockIdx.x*blockDim.x + threadIdx.x; t < NE; t += gridDim.x*blockDim.x) {
    int src = eidx[t], dst = eidx[NE + t];
    int ls = (lig[src] == 1), ld = (lig[dst] == 1);
    ET[t] = ls ? (ld ? 0 : 1) : (ld ? 2 : 3);
    float dx = x[dst*3+0]-x[src*3+0], dy = x[dst*3+1]-x[src*3+1], dz = x[dst*3+2]-x[src*3+2];
    float d = sqrtf(dx*dx + dy*dy + dz*dz);
    float step = logf(11.0f) / 19.0f;
    float acc = ew_b[0], oprev = 0.f;
    for (int j = 0; j < 20; ++j) {
      float oj = (j == 0) ? 0.f : (expf(step*j) - 1.f);
      float w  = (j == 0) ? (expf(step) - 1.f) : (oj - oprev);
      float tt = d - oj;
      acc += expf((-0.5f/(w*w)) * tt*tt) * ew_w[j];
      oprev = oj;
    }
    EW[t] = 1.f / (1.f + expf(-acc));
  }
}

// ---------------- node MLP: q = MLP(h), optional f16 copy of h ----------------

__global__ void __launch_bounds__(256)
node_mlp(const float* __restrict__ h_in, MlpW m, float* __restrict__ q_out,
         f16* __restrict__ h16_out, int nrows) {
  __shared__ f16   hb[128*128];
  __shared__ float ybuf[128*128];
  int tid = threadIdx.x, lane = tid & 31, wave = tid >> 5;
  int row0 = blockIdx.x * 128;
  for (int t = tid; t < 128*128; t += 256) {
    int r = t >> 7, c = t & 127, gr = row0 + r;
    float v = (gr < nrows) ? h_in[(size_t)gr*128 + c] : 0.f;
    hb[t] = (f16)v;
    if (h16_out && gr < nrows) h16_out[(size_t)gr*128 + c] = (f16)v;
  }
  __syncthreads();
  gemm_to_y<4,128>(hb, m.w1p, m.b1, ybuf, wave, lane);
  __syncthreads();
  ln_relu_pack(ybuf, m.g, m.bln, tid);
  __syncthreads();
  v8f acc[8];
  gemm2_128((const f16*)ybuf, m.w2p, acc, wave, lane);
  int colb = lane & 15, rowo = wave*16 + ((lane & 16) ? 8 : 0);
#pragma unroll
  for (int nt = 0; nt < 8; ++nt) {
    int col = nt*16 + colb;
    float bb = m.b2[col];
#pragma unroll
    for (int i = 0; i < 8; ++i) {
      int gr = row0 + rowo + i;
      if (gr < nrows) q_out[(size_t)gr*128 + col] = acc[nt][i] + bb;
    }
  }
}

// ---------------- build 128 x KVP f16 KV block in LDS ----------------

__device__ __forceinline__ void build_kv(f16* kvb, float* eww, float* relb,
    const float* x_in, const f16* h16, const float* ewg, const int* etg,
    const int* eidx, const float* s_off, const float* s_coeff,
    int node0, int edge0, int tid) {
  if (tid < 128) {
    int e = tid, ge = edge0 + e;
    int src = eidx[ge];
    int dstn = eidx[NE + ge];
    float dx = x_in[dstn*4+0] - x_in[src*4+0];
    float dy = x_in[dstn*4+1] - x_in[src*4+1];
    float dz = x_in[dstn*4+2] - x_in[src*4+2];
    float d = sqrtf(dx*dx + dy*dy + dz*dz);
    if (relb) { relb[e*4+0] = dx; relb[e*4+1] = dy; relb[e*4+2] = dz; relb[e*4+3] = 0.f; }
    f16* row = kvb + (size_t)e * KVP;
    h8 z8 = (h8)(_Float16)0.0f;
#pragma unroll
    for (int j = 0; j < KVP/8; ++j) ((h8*)row)[j] = z8;
    int et = etg[ge];
    row[et] = (f16)1.0f;
#pragma unroll
    for (int j = 0; j < 20; ++j) {
      float t = d - s_off[j];
      row[4 + et*20 + j] = (f16)__expf(s_coeff[j] * t * t);
    }
    const h8* hd = (const h8*)(h16 + (size_t)dstn*128);
    const h8* hs = (const h8*)(h16 + (size_t)src*128);
    h8* od = (h8*)(row + 88);
    h8* os = (h8*)(row + 216);
#pragma unroll
    for (int j = 0; j < 16; ++j) { od[j] = hd[j]; os[j] = hs[j]; }
    eww[e] = ewg[ge];
  }
}

__device__ __forceinline__ void softmax32(const float* logits, float* alpha, int tid) {
  if (tid < 64) {                       // 4 nodes x 16 heads
    int ln = tid >> 4, hh = tid & 15, b = ln * 32;
    float m = -1e30f;
    for (int j = 0; j < 32; ++j) m = fmaxf(m, logits[(b+j)*16 + hh]);
    float s = 0.f;
    for (int j = 0; j < 32; ++j) {
      float ex = __expf(logits[(b+j)*16 + hh] - m);
      alpha[(b+j)*16 + hh] = ex; s += ex;
    }
    float inv = 1.f / (s + 1e-16f);
    for (int j = 0; j < 32; ++j) alpha[(b+j)*16 + hh] *= inv;
  }
}

// ---------------- X2H edge kernel: 4 dst nodes / block ----------------

__global__ void __launch_bounds__(256)
edge_x2h(const float* __restrict__ x_in, const float* __restrict__ h_in,
         const f16* __restrict__ h16, const float* __restrict__ q,
         const float* __restrict__ ewg, const int* __restrict__ etg,
         const int* __restrict__ eidx, MlpW hk, MlpW hv,
         float* __restrict__ h_out) {
  __shared__ f16   kvb[128*KVP];
  __shared__ float ybuf[128*128];
  __shared__ float logits[128*16];
  __shared__ float alpha[128*16];
  __shared__ float eww[128];
  __shared__ float agg[4*128];
  __shared__ float s_off[20], s_coeff[20];

  int tid = threadIdx.x, lane = tid & 31, wave = tid >> 5;
  int node0 = blockIdx.x * 4, edge0 = node0 * KNN;

  rbf_consts(s_off, s_coeff, tid);
  if (tid < 512) agg[tid] = 0.f;
  __syncthreads();
  build_kv(kvb, eww, nullptr, x_in, h16, ewg, etg, eidx, s_off, s_coeff, node0, edge0, tid);
  __syncthreads();

  // K = MLP_hk(kv)
  gemm_to_y<11,KVP>(kvb, hk.w1p, hk.b1, ybuf, wave, lane);
  __syncthreads();
  ln_relu_pack(ybuf, hk.g, hk.bln, tid);
  __syncthreads();
  v8f acc[8];
  gemm2_128((const f16*)ybuf, hk.w2p, acc, wave, lane);
  __syncthreads();
  {
    int colb = lane & 15, rowo = wave*16 + ((lane & 16) ? 8 : 0);
#pragma unroll
    for (int nt = 0; nt < 8; ++nt) {
      int col = nt*16 + colb;
      float bb = hk.b2[col];
#pragma unroll
      for (int i = 0; i < 8; ++i)
        ybuf[(size_t)(rowo + i)*128 + col] = acc[nt][i] + bb;
    }
  }
  __syncthreads();
  // logits = (q[dst] . k) / sqrt(d_head)
  for (int t = tid; t < 128*16; t += 256) {
    int e = t >> 4, hh = t & 15;
    int dstn = node0 + (e >> 5);
    const float* qq = q + (size_t)dstn*128 + hh*8;
    const float* kk = ybuf + (size_t)e*128 + hh*8;
    float s = 0.f;
#pragma unroll
    for (int d = 0; d < 8; ++d) s += qq[d] * kk[d];
    logits[t] = s * INV_SQRT_DH;
  }
  __syncthreads();
  softmax32(logits, alpha, tid);
  __syncthreads();

  // V = MLP_hv(kv) * e_w ; agg += alpha * V
  gemm_to_y<11,KVP>(kvb, hv.w1p, hv.b1, ybuf, wave, lane);
  __syncthreads();
  ln_relu_pack(ybuf, hv.g, hv.bln, tid);
  __syncthreads();
  gemm2_128((const f16*)ybuf, hv.w2p, acc, wave, lane);
  {
    int colb = lane & 15;
    int rowbase = wave*16 + ((lane & 16) ? 8 : 0);
    int nodeLocal = wave >> 1;                 // 32 edges per node, 16 per wave tile
#pragma unroll
    for (int nt = 0; nt < 8; ++nt) {
      int col = nt*16 + colb;
      int head = col >> 3;
      float bb = hv.b2[col];
      float s = 0.f;
#pragma unroll
      for (int i = 0; i < 8; ++i) {
        int e = rowbase + i;
        s += (acc[nt][i] + bb) * eww[e] * alpha[e*16 + head];
      }
      atomicAdd(&agg[nodeLocal*128 + col], s);
    }
  }
  __syncthreads();
  for (int t = tid; t < 512; t += 256) {
    int ln = t >> 7, c = t & 127;
    int noden = node0 + ln;
    h_out[(size_t)noden*128 + c] = h_in[(size_t)noden*128 + c] + agg[t];
  }
}

// ---------------- H2X edge kernel ----------------

__global__ void __launch_bounds__(256)
edge_h2x(const float* __restrict__ x_in, const f16* __restrict__ h16,
         const float* __restrict__ q2, const float* __restrict__ ewg,
         const int* __restrict__ etg, const int* __restrict__ eidx,
         const float* __restrict__ gen, MlpW xk, MlpW xv,
         float* __restrict__ x_out) {
  __shared__ f16   kvb[128*KVP];
  __shared__ float ybuf[128*128];
  __shared__ float logits[128*16];       // reused as xv-out after softmax
  __shared__ float alpha[128*16];
  __shared__ float eww[128];
  __shared__ float relb[128*4];
  __shared__ float dxs[16];
  __shared__ float s_off[20], s_coeff[20];

  int tid = threadIdx.x, lane = tid & 31, wave = tid >> 5;
  int node0 = blockIdx.x * 4, edge0 = node0 * KNN;

  rbf_consts(s_off, s_coeff, tid);
  if (tid < 16) dxs[tid] = 0.f;
  __syncthreads();
  build_kv(kvb, eww, relb, x_in, h16, ewg, etg, eidx, s_off, s_coeff, node0, edge0, tid);
  __syncthreads();

  // K2 = MLP_xk(kv2)
  gemm_to_y<11,KVP>(kvb, xk.w1p, xk.b1, ybuf, wave, lane);
  __syncthreads();
  ln_relu_pack(ybuf, xk.g, xk.bln, tid);
  __syncthreads();
  v8f acc[8];
  gemm2_128((const f16*)ybuf, xk.w2p, acc, wave, lane);
  __syncthreads();
  {
    int colb = lane & 15, rowo = wave*16 + ((lane & 16) ? 8 : 0);
#pragma unroll
    for (int nt = 0; nt < 8; ++nt) {
      int col = nt*16 + colb;
      float bb = xk.b2[col];
#pragma unroll
      for (int i = 0; i < 8; ++i)
        ybuf[(size_t)(rowo + i)*128 + col] = acc[nt][i] + bb;
    }
  }
  __syncthreads();
  for (int t = tid; t < 128*16; t += 256) {
    int e = t >> 4, hh = t & 15;
    int dstn = node0 + (e >> 5);
    const float* qq = q2 + (size_t)dstn*128 + hh*8;
    const float* kk = ybuf + (size_t)e*128 + hh*8;
    float s = 0.f;
#pragma unroll
    for (int d = 0; d < 8; ++d) s += qq[d] * kk[d];
    logits[t] = s * INV_SQRT_DH;
  }
  __syncthreads();
  softmax32(logits, alpha, tid);
  __syncthreads();

  // xv: 340 -> 128 -> 16
  gemm_to_y<11,KVP>(kvb, xv.w1p, xv.b1, ybuf, wave, lane);
  __syncthreads();
  ln_relu_pack(ybuf, xv.g, xv.bln, tid);
  __syncthreads();
  v8f acc1 = (v8f)0.0f;
  {
    const f16* zrow = (const f16*)ybuf + (size_t)(wave*16 + (lane & 15)) * 256;
#pragma unroll
    for (int kt = 0; kt < 4; ++kt) {
      v16h a = load_a(zrow + kt*32, lane);
      v16h b = *(const v16h*)(xv.w2p + (size_t)kt*512 + lane*16);
      acc1 = __builtin_amdgcn_wmma_f32_16x16x32_f16(false, a, false, b,
                                                    (short)0, acc1, false, false);
    }
  }
  __syncthreads();
  {
    int colb = lane & 15, rowo = wave*16 + ((lane & 16) ? 8 : 0);
    float bb = xv.b2[colb];
#pragma unroll
    for (int i = 0; i < 8; ++i)
      logits[(rowo + i)*16 + colb] = acc1[i] + bb;    // xv output [128 x 16]
  }
  __syncthreads();
  if (tid < 128) {
    int e = tid;
    float s = 0.f;
#pragma unroll
    for (int h = 0; h < 16; ++h) s += alpha[e*16 + h] * logits[e*16 + h];
    s *= eww[e] * (1.0f / 16.0f);                     // mean over heads
    int ln = e >> 5;
    atomicAdd(&dxs[ln*4+0], s * relb[e*4+0]);
    atomicAdd(&dxs[ln*4+1], s * relb[e*4+1]);
    atomicAdd(&dxs[ln*4+2], s * relb[e*4+2]);
  }
  __syncthreads();
  if (tid < 16) {
    int ln = tid >> 2, a = tid & 3;
    int noden = node0 + ln;
    float v = (a < 3) ? (x_in[noden*4 + a] + dxs[ln*4 + a] * gen[noden]) : 0.f;
    x_out[noden*4 + a] = v;
  }
}

// ---------------- output ----------------

__global__ void finalize(const float* __restrict__ XF, const float* __restrict__ HF,
                         float* __restrict__ out, int out_size) {
  for (int t = blockIdx.x*blockDim.x + threadIdx.x; t < out_size; t += gridDim.x*blockDim.x) {
    if (t < NN*3) out[t] = XF[(t/3)*4 + (t%3)];
    else          out[t] = HF[t - NN*3];
  }
}

// ---------------- host ----------------

extern "C" void kernel_launch(void* const* d_in, const int* in_sizes, int n_in,
                              void* d_out, int out_size, void* d_ws, size_t ws_size,
                              hipStream_t stream) {
  (void)in_sizes; (void)n_in; (void)ws_size;
  const float* x0   = (const float*)d_in[0];
  const float* h0   = (const float*)d_in[1];
  const int*   lig  = (const int*)d_in[3];
  const float* gen  = (const float*)d_in[4];
  const int*   eidx = (const int*)d_in[5];
  const float* ewb  = (const float*)d_in[6];   // params.ew.b  (dict keys sorted: ew < layers; b < w)
  const float* eww_ = (const float*)d_in[7];   // params.ew.w
  // layer leaves: 8 + l*36 + mlp*6 + leaf ; mlp order hk,hq,hv,xk,xq,xv ; leaf b1,b2,bln,g,w1,w2
  auto leaf = [&](int l, int m, int f) -> const float* {
    return (const float*)d_in[8 + l*36 + m*6 + f];
  };

  char* ws = (char*)d_ws;
  size_t off = 0;
  auto alloc = [&](size_t b) -> char* {
    size_t r = (off + 255) & ~(size_t)255;
    off = r + b;
    return ws + r;
  };
  float* XA  = (float*)alloc((size_t)NN*4*4);
  float* XB  = (float*)alloc((size_t)NN*4*4);
  float* HA  = (float*)alloc((size_t)NN*HID*4);
  float* HB  = (float*)alloc((size_t)NN*HID*4);
  f16*   H16 = (f16*)  alloc((size_t)NN*HID*2);
  float* Q   = (float*)alloc((size_t)NN*HID*4);
  float* Q2  = (float*)alloc((size_t)NN*HID*4);
  float* EW  = (float*)alloc((size_t)NE*4);
  int*   ET  = (int*)  alloc((size_t)NE*4);

  MlpW mw[6][6];
  for (int l = 0; l < 6; ++l)
    for (int m = 0; m < 6; ++m) {
      bool isKV = (m == 0 || m == 2 || m == 3 || m == 5);   // hk,hv,xk,xv take the 340-dim KV
      int n2 = (m == 5) ? 16 : 128;
      f16* w1p = (f16*)alloc(isKV ? (size_t)KVP*128*2 : (size_t)128*128*2);
      f16* w2p = (f16*)alloc((size_t)128*n2*2);
      mw[l][m].b1  = leaf(l,m,0); mw[l][m].b2 = leaf(l,m,1);
      mw[l][m].bln = leaf(l,m,2); mw[l][m].g  = leaf(l,m,3);
      mw[l][m].w1p = w1p; mw[l][m].w2p = w2p;
      if (isKV) pack_kv<<<176, 256, 0, stream>>>(leaf(l,m,4), w1p, 128);
      else      pack_plain<<<64, 256, 0, stream>>>(leaf(l,m,4), w1p, 128, 128, 128);
      pack_plain<<<(m==5)?8:64, 256, 0, stream>>>(leaf(l,m,5), w2p, 128, 128, n2);
    }

  init_state<<<2000, 256, 0, stream>>>(x0, h0, XA, HA);
  init_edges<<<500, 256, 0, stream>>>(x0, lig, eidx, eww_, ewb, EW, ET);

  for (int l = 0; l < 6; ++l) {
    float* xi = (l & 1) ? XB : XA; float* xo = (l & 1) ? XA : XB;
    float* hi = (l & 1) ? HB : HA; float* ho = (l & 1) ? HA : HB;
    node_mlp<<<32, 256, 0, stream>>>(hi, mw[l][1], Q,  H16, NN);          // q  = hq(h)
    edge_x2h<<<NN/4, 256, 0, stream>>>(xi, hi, H16, Q, EW, ET, eidx,
                                       mw[l][0], mw[l][2], ho);           // h update
    node_mlp<<<32, 256, 0, stream>>>(ho, mw[l][4], Q2, H16, NN);          // q2 = xq(h_new)
    edge_h2x<<<NN/4, 256, 0, stream>>>(xi, H16, Q2, EW, ET, eidx, gen,
                                       mw[l][3], mw[l][5], xo);           // x update
  }
  finalize<<<2048, 256, 0, stream>>>(XA, HA, (float*)d_out, out_size);
}
